// SelfAttention_63599875719262
// MI455X (gfx1250) — matrix operand order repaired and checked by reading, not compiled, not run on hardware
//
#include <hip/hip_runtime.h>
#include <hip/hip_bf16.h>

// Problem constants
#define NB      8
#define LSEQ    1024
#define DMODEL  768
#define NH      12
#define DKH     64
#define MROWS   (NB * LSEQ)      // 8192
#define D3      (3 * DMODEL)     // 2304

typedef _Float16 h16v __attribute__((ext_vector_type(16)));
typedef _Float16 h8v  __attribute__((ext_vector_type(8)));
typedef float    f8v  __attribute__((ext_vector_type(8)));
typedef unsigned int u32x4 __attribute__((ext_vector_type(4)));
typedef unsigned int u32x8 __attribute__((ext_vector_type(8)));

// ---------------------------------------------------------------------------
// WMMA helper (CDNA5 16x16x32 f16 -> f32)
// ---------------------------------------------------------------------------
__device__ __forceinline__ f8v wmma_f16(h16v a, h16v b, f8v c) {
  return __builtin_amdgcn_wmma_f32_16x16x32_f16(
      /*neg_a=*/false, a, /*neg_b=*/false, b,
      /*c_mod=*/(short)0, c, /*reuse_a=*/false, /*reuse_b=*/false);
}

// A-matrix 16x32 f16 fragment from row-major src (row stride `ld` halfs).
// lo lanes hold k in {k0..k0+7, k0+16..k0+23}; hi lanes hold k+8 of each chunk.
__device__ __forceinline__ h16v load_fragA(const _Float16* base, int ld, int k0, int lane) {
  const int m = lane & 15, g = lane >> 4;
  const _Float16* p = base + (size_t)m * ld + k0 + 8 * g;
  h8v lo = *(const h8v*)p;
  h8v hi = *(const h8v*)(p + 16);
  h16v a;
#pragma unroll
  for (int i = 0; i < 8; ++i) { a[i] = lo[i]; a[i + 8] = hi[i]; }
  return a;
}

// B-matrix 32x16 fragment where B[k][n] = src[n][k] (src row-major, stride ld).
// lane holds column n = lane&15; lo lanes k in {k0..k0+7, k0+16..k0+23}; hi lanes k+8 of each chunk.
__device__ __forceinline__ h16v load_fragB_rows(const _Float16* base, int ld, int k0, int lane) {
  const int n = lane & 15, g = lane >> 4;
  const _Float16* p = base + (size_t)n * ld + k0 + 8 * g;
  h8v lo = *(const h8v*)p;
  h8v hi = *(const h8v*)(p + 16);
  h16v b;
#pragma unroll
  for (int i = 0; i < 8; ++i) { b[i] = lo[i]; b[i + 8] = hi[i]; }
  return b;
}

// ---------------------------------------------------------------------------
// Tensor Data Mover: 2D f16 tile load (global -> LDS), D# per CDNA5 ISA ch.8.
// tile = tile_d1 rows x tile_d0 halfs (contiguous), row stride `stride0` halfs.
// Packed row-major into LDS at lds_byte_addr.  One wave issues; TENSORcnt.
// ---------------------------------------------------------------------------
__device__ __forceinline__ void tdm_load_2d_f16(unsigned lds_byte_addr,
                                                const void* gsrc,
                                                unsigned tile_d0, unsigned tile_d1,
                                                unsigned long long stride0) {
  unsigned long long ga = (unsigned long long)gsrc;
  u32x4 g0;
  g0[0] = 1u;                                   // count=1, user descriptor
  g0[1] = lds_byte_addr;                        // LDS byte address
  g0[2] = (unsigned)ga;                         // global_addr[31:0]
  g0[3] = (unsigned)((ga >> 32) & 0x01ffffffu)  // global_addr[56:32]
          | (2u << 30);                         // type=2 ("image")
  u32x8 g1;
  g1[0] = 1u << 16;                             // data_size=1 (2B); no multicast
  g1[1] = (tile_d0 & 0xffffu) << 16;            // tensor_dim0[15:0]  (== tile_d0)
  g1[2] = (tile_d1 & 0xffffu) << 16;            // tensor_dim1[15:0]  (== tile_d1)
  g1[3] = (tile_d0 & 0xffffu) << 16;            // tile_dim0
  g1[4] = (tile_d1 & 0xffffu);                  // tile_dim1; tile_dim2=0
  g1[5] = (unsigned)stride0;                    // tensor_dim0_stride[31:0]
  g1[6] = (unsigned)((stride0 >> 32) & 0xffffu);// stride[47:32]; dim1_stride=0
  g1[7] = 0u;
  // 2D form: VADDR2/VADDR3 omitted (NULL) per ISA 7.3
  asm volatile("tensor_load_to_lds %0, %1" :: "s"(g0), "s"(g1) : "memory");
}

__device__ __forceinline__ void wait_tensorcnt0() {
#if __has_builtin(__builtin_amdgcn_s_wait_tensorcnt)
  __builtin_amdgcn_s_wait_tensorcnt(0);
#else
  asm volatile("s_wait_tensorcnt 0x0" ::: "memory");
#endif
}

__device__ __forceinline__ unsigned lds_dyn_base() {
#if __has_builtin(__builtin_amdgcn_groupstaticsize)
  return __builtin_amdgcn_groupstaticsize();
#else
  return 0u;
#endif
}

// ---------------------------------------------------------------------------
// Kernel: f32 -> f16 cast / zero
// ---------------------------------------------------------------------------
__global__ void k_cast_f16(_Float16* __restrict__ dst, const float* __restrict__ src, int n) {
  int i = blockIdx.x * blockDim.x + threadIdx.x;
  if (i < n) dst[i] = (_Float16)src[i];
}

__global__ void k_zero_f32(float* __restrict__ dst, int n) {
  int i = blockIdx.x * blockDim.x + threadIdx.x;
  if (i < n) dst[i] = 0.0f;
}

// ---------------------------------------------------------------------------
// Kernel: V transpose.  vT[((b*NH+h)*DKH + j)*LSEQ + l] = V[b,l,h,j]
// (runs after k_qkv_gemm; writes into the dead xh region)
// ---------------------------------------------------------------------------
__global__ void k_transpose_v(_Float16* __restrict__ vT, const _Float16* __restrict__ qkv) {
  int idx = blockIdx.x * 256 + threadIdx.x;   // over NB*NH*DKH*LSEQ = 6291456
  int l  = idx & (LSEQ - 1);
  int j  = (idx >> 10) & (DKH - 1);
  int bh = idx >> 16;                          // 0..95
  int b = bh / NH, h = bh % NH;
  vT[idx] = qkv[(size_t)(b * LSEQ + l) * D3 + 2 * DMODEL + h * DKH + j];
}

// ---------------------------------------------------------------------------
// Kernel: fused QKV projection.  qkv[m][0:768]=Q, [768:1536]=K, [1536:2304]=V
// Block = 128 input rows x 64 output cols.  TDM stages both slabs in LDS:
//   Xs [128][768] f16 = 192 KB,  Ws [64][768] f16 = 96 KB  (288 KB total).
// Inner loop is ds_load -> v_wmma only; per wave 16x64 strip.
// Grid: 64 row-blocks x 36 col-blocks = 2304.
// ---------------------------------------------------------------------------
__global__ void k_qkv_gemm(_Float16* __restrict__ qkv,
                           const _Float16* __restrict__ xh,
                           const _Float16* __restrict__ wqkv,
                           const float* __restrict__ bq,
                           const float* __restrict__ bk,
                           const float* __restrict__ bv) {
  extern __shared__ char smem[];
  _Float16* Xs = (_Float16*)smem;                 // [128][768]
  _Float16* Ws = (_Float16*)(smem + 196608);      // [64][768]

  const int lane = threadIdx.x & 31;
  const int wave = threadIdx.x >> 5;
  const int ni = blockIdx.x / 64;                 // 0..35  (output col block)
  const int mblk = blockIdx.x % 64;               // 0..63  (input row block)
  const int m0_blk = mblk * 128;
  const int m0 = m0_blk + wave * 16;              // this wave's 16 rows
  const int n0 = ni * 64;                         // block-shared 64 cols

  if (wave == 0) {
    const unsigned base = lds_dyn_base();
    tdm_load_2d_f16(base, xh + (size_t)m0_blk * DMODEL, DMODEL, 128, DMODEL);
    tdm_load_2d_f16(base + 196608, wqkv + (size_t)n0 * DMODEL, DMODEL, 64, DMODEL);
    wait_tensorcnt0();
  }
  __syncthreads();

  f8v acc[4];
#pragma unroll
  for (int j = 0; j < 4; ++j)
#pragma unroll
    for (int r = 0; r < 8; ++r) acc[j][r] = 0.0f;

  const _Float16* abase = Xs + (size_t)(wave * 16) * DMODEL;
  for (int kc = 0; kc < DMODEL; kc += 32) {
    h16v a = load_fragA(abase, DMODEL, kc, lane);
    h16v b[4];
#pragma unroll
    for (int j = 0; j < 4; ++j)
      b[j] = load_fragB_rows(Ws + (size_t)(16 * j) * DMODEL, DMODEL, kc, lane);
#pragma unroll
    for (int j = 0; j < 4; ++j) acc[j] = wmma_f16(a, b[j], acc[j]);
  }

  const int g = lane >> 4;
#pragma unroll
  for (int j = 0; j < 4; ++j) {
    const int n = n0 + 16 * j + (lane & 15);
    float bias = (n < DMODEL) ? bq[n]
               : (n < 2 * DMODEL) ? bk[n - DMODEL] : bv[n - 2 * DMODEL];
    _Float16* outp = qkv + (size_t)(m0 + 8 * g) * D3 + n;
#pragma unroll
    for (int r = 0; r < 8; ++r)
      outp[(size_t)r * D3] = (_Float16)(acc[j][r] + bias);
  }
}

// ---------------------------------------------------------------------------
// Kernel: attention for one (b, head, 32-query tile).
//   TDM loads whole K slice [1024x64] f16 into LDS
//   scores (WMMA, B-frags from LDS) -> LDS f32 [32][1024]
//   exact softmax -> probs f16 in LDS [32][1024]
//   column sums -> global f32 atomics (pooling weights)
//   attn @ V (WMMA, A from LDS, B contiguous from vT, double-buffered)
// LDS: S 131072 + P 65536 + K 131072 = 327680 B (= 320 KB max);
// softmax partials alias the K region (K dead after scores).
// ---------------------------------------------------------------------------
__global__ void k_attn(_Float16* __restrict__ ctx, float* __restrict__ colsum,
                       const _Float16* __restrict__ qkv,
                       const _Float16* __restrict__ vT) {
  extern __shared__ char smem[];
  float*    S  = (float*)smem;                       // [32][1024]
  _Float16* P  = (_Float16*)(smem + 131072);         // [32][1024] f16
  _Float16* Kb = (_Float16*)(smem + 196608);         // [1024][64] f16 (K slice)
  float*  part = (float*)(smem + 196608);            // [32][8] alias over Kb

  const int tid  = threadIdx.x;
  const int lane = tid & 31;
  const int wave = tid >> 5;
  const int qt = blockIdx.x & 31;
  const int hh = (blockIdx.x >> 5) % NH;
  const int bb = blockIdx.x / (32 * NH);
  const int q0 = qt * 32;

  const _Float16* qbase = qkv + (size_t)(bb * LSEQ + q0) * D3 + hh * DKH;
  const _Float16* kbase = qkv + (size_t)(bb * LSEQ) * D3 + DMODEL + hh * DKH;

  // ---- kick off TDM for the K slice, overlap with Q fragment loads ----
  if (wave == 0)
    tdm_load_2d_f16(lds_dyn_base() + 196608, kbase, DKH, LSEQ,
                    (unsigned long long)D3);

  const int mi = wave & 1;                           // 16-row q sub-tile
  h16v a0 = load_fragA(qbase + (size_t)mi * 16 * D3, D3, 0, lane);
  h16v a1 = load_fragA(qbase + (size_t)mi * 16 * D3, D3, 32, lane);

  if (wave == 0) wait_tensorcnt0();
  __syncthreads();

  // ---- scores = Q K^T / sqrt(dk), K fragments from LDS ----
  {
    const float scale = 0.125f;                      // 1/sqrt(64)
    const int g = lane >> 4, n = lane & 15;
    for (int t = (wave >> 1); t < 64; t += 4) {      // 16 key tiles per wave
      const int n0 = t * 16;
      f8v acc;
#pragma unroll
      for (int r = 0; r < 8; ++r) acc[r] = 0.0f;
      h16v b0 = load_fragB_rows(Kb + (size_t)n0 * DKH, DKH, 0, lane);
      h16v b1 = load_fragB_rows(Kb + (size_t)n0 * DKH, DKH, 32, lane);
      acc = wmma_f16(a0, b0, acc);
      acc = wmma_f16(a1, b1, acc);
#pragma unroll
      for (int r = 0; r < 8; ++r)
        S[(mi * 16 + r + 8 * g) * 1024 + n0 + n] = acc[r] * scale;
    }
  }
  __syncthreads();

  // ---- exact softmax over full rows in LDS (part aliases dead K region) ----
  {
    const int r = tid >> 3, sub = tid & 7;           // 8 threads per row
    float* srow = S + r * 1024;
    const int c0 = sub * 128;
    float m = -1e30f;
    for (int c = 0; c < 128; ++c) m = fmaxf(m, srow[c0 + c]);
    part[r * 8 + sub] = m;
    __syncthreads();
    float rowmax = part[r * 8];
#pragma unroll
    for (int i = 1; i < 8; ++i) rowmax = fmaxf(rowmax, part[r * 8 + i]);
    __syncthreads();
    float s = 0.0f;
    for (int c = 0; c < 128; ++c) {
      float e = __expf(srow[c0 + c] - rowmax);
      srow[c0 + c] = e;
      s += e;
    }
    part[r * 8 + sub] = s;
    __syncthreads();
    float rowsum = 0.0f;
#pragma unroll
    for (int i = 0; i < 8; ++i) rowsum += part[r * 8 + i];
    const float inv = 1.0f / rowsum;
    _Float16* prow = P + r * 1024;
    for (int c = 0; c < 128; ++c)
      prow[c0 + c] = (_Float16)(srow[c0 + c] * inv);
  }
  __syncthreads();

  // ---- accumulate attention column sums (pooling weights) ----
  for (int c = tid; c < 1024; c += 256) {
    float s = 0.0f;
#pragma unroll 8
    for (int r = 0; r < 32; ++r) s += (float)P[r * 1024 + c];
    atomicAdd(&colsum[bb * 1024 + c], s);
  }

  // ---- context = attn @ V  (B contiguous from vT, double-buffered) ----
  {
    const int ni = wave >> 1;
    const int n0 = ni * 16;
    f8v acc;
#pragma unroll
    for (int r = 0; r < 8; ++r) acc[r] = 0.0f;
    const _Float16* pa  = P + mi * 16 * 1024;
    const _Float16* vtb = vT + ((size_t)(bb * NH + hh) * DKH + n0) * LSEQ;
    h16v bcur = load_fragB_rows(vtb, LSEQ, 0, lane);
    for (int kc = 0; kc < LSEQ - 32; kc += 32) {
      h16v bnxt = load_fragB_rows(vtb, LSEQ, kc + 32, lane);
      h16v a = load_fragA(pa, 1024, kc, lane);       // ds_load from LDS
      acc = wmma_f16(a, bcur, acc);
      bcur = bnxt;
    }
    {
      h16v a = load_fragA(pa, 1024, LSEQ - 32, lane);
      acc = wmma_f16(a, bcur, acc);
    }
    const int g = lane >> 4, n = lane & 15;
    _Float16* outp = ctx + (size_t)(bb * LSEQ + q0 + mi * 16 + 8 * g) * DMODEL
                     + hh * DKH + n0 + n;
#pragma unroll
    for (int r = 0; r < 8; ++r) outp[(size_t)r * DMODEL] = (_Float16)acc[r];
  }
}

// ---------------------------------------------------------------------------
// Kernel: out-projection + residual + LayerNorm, 16 rows per block.
// TDM stages the block-shared ctx slab [16][768] f16 (24 KB) in LDS;
// B-fragments from global in groups of 3 (acc 48 + b 24 + a 8 live regs).
// LDS: 24576 + 49152 + 2048 = 75776 B.
// ---------------------------------------------------------------------------
__global__ void k_proj_ln(float* __restrict__ y, const _Float16* __restrict__ ctx,
                          const _Float16* __restrict__ woh, const float* __restrict__ bo,
                          const float* __restrict__ x, const float* __restrict__ gamma,
                          const float* __restrict__ beta) {
  extern __shared__ char smem[];
  _Float16* Cs = (_Float16*)smem;                   // [16][768]
  float* rowbuf = (float*)(smem + 24576);           // [16][768]
  float* pa = (float*)(smem + 24576 + 49152);       // [256]
  float* pb = pa + 256;                             // [256]
  const int tid = threadIdx.x, lane = tid & 31, wave = tid >> 5;
  const int m0 = blockIdx.x * 16;

  if (wave == 0) {
    tdm_load_2d_f16(lds_dyn_base(), ctx + (size_t)m0 * DMODEL, DMODEL, 16, DMODEL);
    wait_tensorcnt0();
  }
  __syncthreads();

  f8v acc[6];
#pragma unroll
  for (int i = 0; i < 6; ++i)
#pragma unroll
    for (int r = 0; r < 8; ++r) acc[i][r] = 0.0f;

  for (int kc = 0; kc < DMODEL; kc += 32) {
    h16v a = load_fragA(Cs, DMODEL, kc, lane);
    h16v b3[3];
#pragma unroll
    for (int i = 0; i < 3; ++i)
      b3[i] = load_fragB_rows(woh + (size_t)((wave * 6 + i) * 16) * DMODEL, DMODEL, kc, lane);
#pragma unroll
    for (int i = 0; i < 3; ++i) acc[i] = wmma_f16(a, b3[i], acc[i]);
#pragma unroll
    for (int i = 0; i < 3; ++i)
      b3[i] = load_fragB_rows(woh + (size_t)((wave * 6 + i + 3) * 16) * DMODEL, DMODEL, kc, lane);
#pragma unroll
    for (int i = 0; i < 3; ++i) acc[i + 3] = wmma_f16(a, b3[i], acc[i + 3]);
  }

  {
    const int g = lane >> 4, n = lane & 15;
#pragma unroll
    for (int i = 0; i < 6; ++i) {
      const int n0 = (wave * 6 + i) * 16;
#pragma unroll
      for (int r = 0; r < 8; ++r)
        rowbuf[(r + 8 * g) * 768 + n0 + n] = acc[i][r];
    }
  }
  __syncthreads();

  const int r = tid >> 4, sub = tid & 15, c0 = sub * 48;  // 16 threads/row
  float* rb = rowbuf + r * 768;
  const float* xrow = x + (size_t)(m0 + r) * DMODEL;
  float lsum = 0.0f;
  for (int c = 0; c < 48; ++c) {
    float t = rb[c0 + c] + bo[c0 + c] + xrow[c0 + c];
    rb[c0 + c] = t;
    lsum += t;
  }
  pa[r * 16 + sub] = lsum;
  __syncthreads();
  float mean = 0.0f;
#pragma unroll
  for (int i = 0; i < 16; ++i) mean += pa[r * 16 + i];
  mean *= (1.0f / 768.0f);
  float lvar = 0.0f;
  for (int c = 0; c < 48; ++c) {
    float d = rb[c0 + c] - mean;
    lvar += d * d;
  }
  pb[r * 16 + sub] = lvar;
  __syncthreads();
  float var = 0.0f;
#pragma unroll
  for (int i = 0; i < 16; ++i) var += pb[r * 16 + i];
  var *= (1.0f / 768.0f);
  const float rstd = rsqrtf(var + 1e-5f);
  float* yrow = y + (size_t)(m0 + r) * DMODEL;
  for (int c = 0; c < 48; ++c)
    yrow[c0 + c] = (rb[c0 + c] - mean) * rstd * gamma[c0 + c] + beta[c0 + c];
}

// ---------------------------------------------------------------------------
// Kernel: attention-weighted pooling.  One block per batch element.
// ---------------------------------------------------------------------------
__global__ void k_pool(float* __restrict__ out, const float* __restrict__ colsum,
                       const float* __restrict__ y) {
  __shared__ float cs[1024];
  __shared__ float red[256];
  const int b = blockIdx.x, tid = threadIdx.x;
  for (int i = tid; i < 1024; i += 256) cs[i] = colsum[b * 1024 + i];
  __syncthreads();
  float l = cs[tid] + cs[tid + 256] + cs[tid + 512] + cs[tid + 768];
  red[tid] = l;
  __syncthreads();
  for (int s = 128; s > 0; s >>= 1) {
    if (tid < s) red[tid] += red[tid + s];
    __syncthreads();
  }
  const float total = red[0];
  const float denom = total * (1.0f / (NH * LSEQ)) + 1e-8f;
  const float coeff = (1.0f / (NH * LSEQ)) / denom;
  for (int d = tid; d < DMODEL; d += 256) {
    float acc = 0.0f;
    const float* yp = y + (size_t)b * LSEQ * DMODEL + d;
    for (int l2 = 0; l2 < LSEQ; ++l2)
      acc += cs[l2] * yp[(size_t)l2 * DMODEL];
    out[b * DMODEL + d] = acc * coeff;
  }
}

// ---------------------------------------------------------------------------
// Host-side launcher
// ---------------------------------------------------------------------------
extern "C" void kernel_launch(void* const* d_in, const int* in_sizes, int n_in,
                              void* d_out, int out_size, void* d_ws, size_t ws_size,
                              hipStream_t stream) {
  const float* x     = (const float*)d_in[0];
  const float* Wq    = (const float*)d_in[1];
  const float* bq    = (const float*)d_in[2];
  const float* Wk    = (const float*)d_in[3];
  const float* bk    = (const float*)d_in[4];
  const float* Wv    = (const float*)d_in[5];
  const float* bv    = (const float*)d_in[6];
  const float* Wo    = (const float*)d_in[7];
  const float* bo    = (const float*)d_in[8];
  const float* gamma = (const float*)d_in[9];
  const float* beta  = (const float*)d_in[10];
  float* out = (float*)d_out;

  char* ws = (char*)d_ws;
  _Float16* xh   = (_Float16*)(ws + 0);          // 12,582,912 B (later reused as vT)
  _Float16* wqkv = (_Float16*)(ws + 12582912);   //  3,538,944 B
  _Float16* woh  = (_Float16*)(ws + 16121856);   //  1,179,648 B
  _Float16* qkv  = (_Float16*)(ws + 17301504);   // 37,748,736 B
  _Float16* ctx  = (_Float16*)(ws + 55050240);   // 12,582,912 B
  float*    y    = (float*)(ws + 67633152);      // 25,165,824 B
  float*    csum = (float*)(ws + 92798976);      //     32,768 B  (total ~88.5 MB)
  _Float16* vT   = xh;                           // alias: xh dead after k_qkv_gemm

  const int nxh = MROWS * DMODEL;
  const int nw  = DMODEL * DMODEL;
  k_cast_f16<<<(nxh + 255) / 256, 256, 0, stream>>>(xh, x, nxh);
  k_cast_f16<<<(nw + 255) / 256, 256, 0, stream>>>(wqkv, Wq, nw);
  k_cast_f16<<<(nw + 255) / 256, 256, 0, stream>>>(wqkv + nw, Wk, nw);
  k_cast_f16<<<(nw + 255) / 256, 256, 0, stream>>>(wqkv + 2 * nw, Wv, nw);
  k_cast_f16<<<(nw + 255) / 256, 256, 0, stream>>>(woh, Wo, nw);
  k_zero_f32<<<(NB * LSEQ + 255) / 256, 256, 0, stream>>>(csum, NB * LSEQ);

  k_qkv_gemm<<<2304, 256, 294912, stream>>>(qkv, xh, wqkv, bq, bk, bv);
  k_transpose_v<<<(NB * NH * DKH * LSEQ) / 256, 256, 0, stream>>>(vT, qkv);
  k_attn<<<NB * NH * 32, 256, 327680, stream>>>(ctx, csum, qkv, vT);
  k_proj_ln<<<MROWS / 16, 256, 75776, stream>>>(y, ctx, woh, bo, x, gamma, beta);
  k_pool<<<NB, 256, 0, stream>>>(out, csum, y);
}
